// CausalSelfAttention_27367531610553
// MI455X (gfx1250) — matrix operand order repaired and checked by reading, not compiled, not run on hardware
//
#include <hip/hip_runtime.h>

typedef __attribute__((ext_vector_type(16))) _Float16 v16h;
typedef __attribute__((ext_vector_type(8)))  _Float16 v8h;
typedef __attribute__((ext_vector_type(4)))  _Float16 v4h;
typedef __attribute__((ext_vector_type(8)))  float    v8f;

__device__ __forceinline__ v16h join8(v8h lo, v8h hi8) {
  v16h a;
#pragma unroll
  for (int j = 0; j < 8; ++j) { a[j] = lo[j]; a[8 + j] = hi8[j]; }
  return a;
}

__device__ __forceinline__ v8f wmma16(v16h a, v16h b, v8f c) {
  return __builtin_amdgcn_wmma_f32_16x16x32_f16(false, a, false, b, (short)0, c,
                                                false, false);
}

// ---------- CDNA5 async global->LDS (ASYNCcnt) ----------
__device__ __forceinline__ unsigned lds_off(const void* p) {
  return (unsigned)(uintptr_t)p;   // low 32 bits of generic LDS pointer = LDS byte offset
}
__device__ __forceinline__ void async_ld_b128(unsigned lds, const void* g) {
  asm volatile("global_load_async_to_lds_b128 %0, %1, off"
               :: "v"(lds), "v"(g) : "memory");
}
__device__ __forceinline__ void wait_async0() {
  asm volatile("s_wait_asynccnt 0" ::: "memory");
}

// ---------- DPP16 half-wave (16-lane) reductions, pure VALU ----------
template <int CTRL>
__device__ __forceinline__ float dpp_mov_f(float v) {
  return __int_as_float(
      __builtin_amdgcn_mov_dpp(__float_as_int(v), CTRL, 0xf, 0xf, true));
}
__device__ __forceinline__ float red_max16(float v) {
  v = fmaxf(v, dpp_mov_f<0xB1>(v));    // quad_perm [1,0,3,2]  (xor 1)
  v = fmaxf(v, dpp_mov_f<0x4E>(v));    // quad_perm [2,3,0,1]  (xor 2)
  v = fmaxf(v, dpp_mov_f<0x141>(v));   // row_half_mirror      (xor 7)
  v = fmaxf(v, dpp_mov_f<0x140>(v));   // row_mirror           (xor 15)
  return v;
}
__device__ __forceinline__ float red_sum16(float v) {
  v += dpp_mov_f<0xB1>(v);
  v += dpp_mov_f<0x4E>(v);
  v += dpp_mov_f<0x141>(v);
  v += dpp_mov_f<0x140>(v);
  return v;
}

// ---------------- f32 -> f16 convert, 4 elems/thread ----------------
__global__ void cvt_f32_f16(const float* __restrict__ in,
                            _Float16* __restrict__ out, int n4) {
  int i = blockIdx.x * blockDim.x + threadIdx.x;
  if (i < n4) {
    float4 v = ((const float4*)in)[i];
    v4h o;
    o[0] = (_Float16)v.x; o[1] = (_Float16)v.y;
    o[2] = (_Float16)v.z; o[3] = (_Float16)v.w;
    ((v4h*)out)[i] = o;
  }
}

// ---------------- tiled f16 WMMA GEMM: C[M,N] = A[M,K]*B[K,N] + bias ----------------
// double-buffered LDS, async staging pipelined one K-step ahead
template <typename OutT>
__global__ void __launch_bounds__(256)
gemm_f16(const _Float16* __restrict__ A, const _Float16* __restrict__ Bm,
         const float* __restrict__ bias, OutT* __restrict__ C,
         int M, int N, int K) {
  constexpr int BM = 128, BN = 128, BK = 32;
  __shared__ _Float16 As[2][BM][BK + 8];
  __shared__ _Float16 Bs[2][BK][BN + 8];
  const int tid  = threadIdx.x;
  const int lane = tid & 31, wave = tid >> 5;
  const int hi   = lane >> 4, l16 = lane & 15;
  const int wm   = wave & 3, wn = wave >> 2;      // 4 waves over M, 2 over N
  const int m0   = blockIdx.y * BM, n0 = blockIdx.x * BN;

  // per-thread staging coordinates
  const int ar = tid >> 1, ah = (tid & 1) * 16;     // A: 2 thr/row, 16-half each
  const int br = tid >> 3, bseg = (tid & 7) * 16;   // B: 8 thr/row, 16-seg each
  const unsigned adst[2] = { lds_off(&As[0][ar][ah]),   lds_off(&As[1][ar][ah]) };
  const unsigned bdst[2] = { lds_off(&Bs[0][br][bseg]), lds_off(&Bs[1][br][bseg]) };

  auto issue = [&](int buf, int k0) {
    const _Float16* asrc = A + (size_t)(m0 + ar) * K + k0 + ah;
    const _Float16* bsrc = Bm + (size_t)(k0 + br) * N + n0 + bseg;
    async_ld_b128(adst[buf],      asrc);
    async_ld_b128(adst[buf] + 16, asrc + 8);
    async_ld_b128(bdst[buf],      bsrc);
    async_ld_b128(bdst[buf] + 16, bsrc + 8);
  };

  v8f acc[2][4] = {};
  issue(0, 0);
  int cur = 0;
  for (int k0 = 0; k0 < K; k0 += BK, cur ^= 1) {
    wait_async0();        // my loads for buf[cur] (and any in flight) done
    __syncthreads();      // everyone's done; everyone finished reading buf[cur^1]
    if (k0 + BK < K) issue(cur ^ 1, k0 + BK);

    v16h af[2], bf[4];
#pragma unroll
    for (int mi = 0; mi < 2; ++mi) {
      const _Float16* rp = &As[cur][wm * 32 + mi * 16 + l16][0];
      af[mi] = join8(*(const v8h*)(rp + 16 * hi), *(const v8h*)(rp + 16 * hi + 8));
    }
#pragma unroll
    for (int ni = 0; ni < 4; ++ni) {
      const int col = wn * 64 + ni * 16 + l16;
      v16h b;
#pragma unroll
      for (int t = 0; t < 16; ++t) b[t] = Bs[cur][16 * hi + t][col];
      bf[ni] = b;
    }
#pragma unroll
    for (int mi = 0; mi < 2; ++mi)
#pragma unroll
      for (int ni = 0; ni < 4; ++ni)
        acc[mi][ni] = wmma16(af[mi], bf[ni], acc[mi][ni]);
  }
#pragma unroll
  for (int mi = 0; mi < 2; ++mi)
#pragma unroll
    for (int ni = 0; ni < 4; ++ni) {
      const int col = n0 + wn * 64 + ni * 16 + l16;
      const float bv = bias[col];
#pragma unroll
      for (int r = 0; r < 8; ++r) {
        const int row = m0 + wm * 32 + mi * 16 + r + 8 * hi;
        C[(size_t)row * N + col] = (OutT)(acc[mi][ni][r] + bv);
      }
    }
}

// ---------------- flash attention: 128 q-rows per block, one (b,h) ----------------
// K/V double-buffered in LDS, async staging pipelined one key-block ahead
__global__ void __launch_bounds__(256)
flash_attn(const _Float16* __restrict__ qkv, _Float16* __restrict__ attn) {
  constexpr int T = 2048, C3 = 3072, Cc = 1024, D = 64;
  __shared__ _Float16 Ks[2][32][D + 8];
  __shared__ _Float16 Vs[2][32][D + 8];
  __shared__ _Float16 Ps[8][16][40];     // wave-private P re-layout buffers
  const int tid  = threadIdx.x, lane = tid & 31, wave = tid >> 5;
  const int hi   = lane >> 4, l16 = lane & 15;
  const int q0   = blockIdx.x * 128;
  const int b    = blockIdx.y >> 4, h = blockIdx.y & 15;
  const _Float16* qb = qkv + (size_t)b * T * C3 + h * D;
  const _Float16* kb = qb + Cc;
  const _Float16* vb = qb + 2 * Cc;

  // per-wave Q A-fragments (rows q0 + wave*16 + 0..15), pre-scaled by 1/sqrt(D)
  v16h qf[2];
  {
    const _Float16* rp = qb + (size_t)(q0 + wave * 16 + l16) * C3;
#pragma unroll
    for (int d0 = 0; d0 < 2; ++d0) {
      qf[d0] = join8(*(const v8h*)(rp + d0 * 32 + 16 * hi),
                     *(const v8h*)(rp + d0 * 32 + 16 * hi + 8));
      qf[d0] = qf[d0] * (_Float16)0.125f;   // exact (power of two)
    }
  }

  float mrow[8], lrow[8];
  v8f o[4] = {};
#pragma unroll
  for (int r = 0; r < 8; ++r) { mrow[r] = -1e30f; lrow[r] = 0.f; }

  // per-thread K/V staging coordinates (32 keys x 64 dims, 16B per thread each)
  const int kr = tid >> 3, kseg = (tid & 7) * 8;
  const unsigned kdst[2] = { lds_off(&Ks[0][kr][kseg]), lds_off(&Ks[1][kr][kseg]) };
  const unsigned vdst[2] = { lds_off(&Vs[0][kr][kseg]), lds_off(&Vs[1][kr][kseg]) };

  auto issueKV = [&](int buf, int j) {
    size_t goff = (size_t)(j * 32 + kr) * C3 + kseg;
    async_ld_b128(kdst[buf], kb + goff);
    async_ld_b128(vdst[buf], vb + goff);
  };

  const int nblk = q0 / 32 + 4;          // causal: only key blocks <= q tile
  issueKV(0, 0);
  int cur = 0;
  for (int j = 0; j < nblk; ++j, cur ^= 1) {
    wait_async0();
    __syncthreads();
    if (j + 1 < nblk) issueKV(cur ^ 1, j + 1);

    // S(16x32) = Qs * K^T  (two 16-col subtiles, two k-steps of 32)
    v8f s[2] = {};
#pragma unroll
    for (int n = 0; n < 2; ++n)
#pragma unroll
      for (int d0 = 0; d0 < 2; ++d0) {
        const _Float16* rp = &Ks[cur][n * 16 + l16][d0 * 32 + 16 * hi];
        v16h bfr = join8(*(const v8h*)rp, *(const v8h*)(rp + 8));
        s[n] = wmma16(qf[d0], bfr, s[n]);
      }

    // online softmax (row = r + 8*hi lives in lanes hi*16..hi*16+15)
#pragma unroll
    for (int r = 0; r < 8; ++r) {
      const int qi = q0 + wave * 16 + r + 8 * hi;
      float s0 = s[0][r];
      float s1 = s[1][r];
      if (j * 32 + l16 > qi)      s0 = -1e30f;
      if (j * 32 + 16 + l16 > qi) s1 = -1e30f;
      const float mx   = red_max16(fmaxf(s0, s1));
      const float mnew = fmaxf(mrow[r], mx);
      const float p0 = __expf(s0 - mnew), p1 = __expf(s1 - mnew);
      const float rs = red_sum16(p0 + p1);
      const float alpha = __expf(mrow[r] - mnew);
      lrow[r] = lrow[r] * alpha + rs;
      mrow[r] = mnew;
#pragma unroll
      for (int n4 = 0; n4 < 4; ++n4) o[n4][r] *= alpha;
      Ps[wave][r + 8 * hi][l16]      = (_Float16)p0;
      Ps[wave][r + 8 * hi][16 + l16] = (_Float16)p1;
    }
    asm volatile("s_wait_dscnt 0" ::: "memory");   // P store->load, same wave

    // O(16x64) += P(16x32) * V(32x64)
    v16h pf;
    {
      const _Float16* rp = &Ps[wave][l16][0];
      pf = join8(*(const v8h*)(rp + 16 * hi), *(const v8h*)(rp + 16 * hi + 8));
    }
#pragma unroll
    for (int n4 = 0; n4 < 4; ++n4) {
      v16h vf;
#pragma unroll
      for (int t = 0; t < 16; ++t) vf[t] = Vs[cur][16 * hi + t][n4 * 16 + l16];
      o[n4] = wmma16(pf, vf, o[n4]);
    }
  }

  // normalize and write f16 attention output [B*T, C]
#pragma unroll
  for (int n4 = 0; n4 < 4; ++n4)
#pragma unroll
    for (int r = 0; r < 8; ++r) {
      const int qi = q0 + wave * 16 + r + 8 * hi;
      attn[(size_t)(b * T + qi) * Cc + h * D + n4 * 16 + l16] =
          (_Float16)(o[n4][r] / lrow[r]);
    }
}

extern "C" void kernel_launch(void* const* d_in, const int* in_sizes, int n_in,
                              void* d_out, int out_size, void* d_ws, size_t ws_size,
                              hipStream_t stream) {
  (void)in_sizes; (void)n_in; (void)out_size; (void)ws_size;
  const float* x      = (const float*)d_in[0];
  const float* W_attn = (const float*)d_in[1];
  const float* b_attn = (const float*)d_in[2];
  const float* W_proj = (const float*)d_in[3];
  const float* b_proj = (const float*)d_in[4];
  float* out = (float*)d_out;

  constexpr int Bb = 2, T = 2048, C = 1024, H = 16;
  const size_t M = (size_t)Bb * T;                  // 4096
  _Float16* xh   = (_Float16*)d_ws;                 // 4096x1024
  _Float16* Wah  = xh   + M * C;                    // 1024x3072
  _Float16* Wph  = Wah  + (size_t)C * 3 * C;        // 1024x1024
  _Float16* qkvh = Wph  + (size_t)C * C;            // 4096x3072
  _Float16* atth = qkvh + M * 3 * C;                // 4096x1024

  const size_t nX = M * C, nWa = (size_t)C * 3 * C, nWp = (size_t)C * C;
  cvt_f32_f16<<<dim3((unsigned)(nX  / 4 / 256)), 256, 0, stream>>>(x,      xh,  (int)(nX  / 4));
  cvt_f32_f16<<<dim3((unsigned)(nWa / 4 / 256)), 256, 0, stream>>>(W_attn, Wah, (int)(nWa / 4));
  cvt_f32_f16<<<dim3((unsigned)(nWp / 4 / 256)), 256, 0, stream>>>(W_proj, Wph, (int)(nWp / 4));

  // qkv[4096,3072] = x * W_attn + b_attn
  gemm_f16<_Float16><<<dim3(3 * C / 128, (unsigned)(M / 128)), 256, 0, stream>>>(
      xh, Wah, b_attn, qkvh, (int)M, 3 * C, C);

  // flash attention -> atth[4096,1024]
  flash_attn<<<dim3(T / 128, Bb * H), 256, 0, stream>>>(qkvh, atth);

  // out[4096,1024] = atth * W_proj + b_proj  (f32 output)
  gemm_f16<float><<<dim3(C / 128, (unsigned)(M / 128)), 256, 0, stream>>>(
      atth, Wph, b_proj, out, (int)M, C, C);
}
